// MicroBiMambaBackbone_46248207843464
// MI455X (gfx1250) — compile-verified, matrix operand order checked by reading, not run on hardware
//
#include <hip/hip_runtime.h>
#include <hip/hip_bf16.h>

// ---------------------------------------------------------------------------
// MicroBiMambaBackbone on MI455X (gfx1250, wave32):
//   WMMA bf16 GEMMs + async global->LDS (ASYNCcnt) + TDM weight tiles
//   (TENSORcnt) + chunk-parallel associative selective scan.
// ---------------------------------------------------------------------------

#define BB   32
#define LL   1024
#define DD   256
#define DI   512
#define SS   16
#define DTR  16
#define NL   4
#define OUTD 128
#define MM   (BB * LL)        // 32768 rows
#define NCH  32               // scan chunks
#define CLEN 32               // steps per chunk

typedef __attribute__((ext_vector_type(16))) __bf16   bf16x16;
typedef __attribute__((ext_vector_type(8)))  float    floatx8;
typedef __attribute__((ext_vector_type(4)))  unsigned v4u;
typedef __attribute__((ext_vector_type(8)))  int      v8i;
typedef __attribute__((ext_vector_type(4)))  int      v4i;

static __device__ __forceinline__ __bf16 f2bf(float f) {
    union { float f; unsigned u; } a; a.f = f;
    unsigned r = a.u + 0x7fffu + ((a.u >> 16) & 1u);
    union { unsigned short u; __bf16 b; } o; o.u = (unsigned short)(r >> 16);
    return o.b;
}

// Async DMA: global -> LDS, 16B per lane, tracked by ASYNCcnt.
// LDS operand: low 32 bits of a generic pointer to a __shared__ object.
#define ASYNC_LDCP(lp, gp)                                                  \
    asm volatile("global_load_async_to_lds_b128 %0, %1, off"                \
                 :: "v"((unsigned)(size_t)(lp)),                            \
                    "v"((unsigned long long)(size_t)(gp)) : "memory")
#define WAIT_ASYNC() asm volatile("s_wait_asynccnt 0x0" ::: "memory")

#if __has_builtin(__builtin_amdgcn_tensor_load_to_lds)
#define HAVE_TDM 1
#endif

#if __has_builtin(__builtin_amdgcn_s_wait_tensorcnt)
#define WAIT_TENSOR() __builtin_amdgcn_s_wait_tensorcnt(0)
#else
#define WAIT_TENSOR() asm volatile("s_wait_tensorcnt 0x0" ::: "memory")
#endif

// ---------------------------------------------------------------------------
// bf16 weight conversion
// ---------------------------------------------------------------------------
__global__ void cvt_bf16_kernel(const float* __restrict__ src,
                                __bf16* __restrict__ dst, int n) {
    int i = blockIdx.x * 256 + threadIdx.x;
    if (i < n) dst[i] = f2bf(src[i]);
}

// xproj (NL,48,512) -> zero-padded (NL,64,512) bf16
__global__ void cvt_xproj_kernel(const float* __restrict__ src,
                                 __bf16* __restrict__ dst) {
    int i = blockIdx.x * 256 + threadIdx.x;      // NL*64*512
    int k = i & 511;
    int n = (i >> 9) & 63;
    int l = i >> 15;
    dst[i] = (n < 48) ? f2bf(src[(l * 48 + n) * 512 + k]) : f2bf(0.0f);
}

// ---------------------------------------------------------------------------
// Embedding: cont-linear + LN + GELU + dir_emb + pos-enc (+ flipped copy).
// ---------------------------------------------------------------------------
__global__ __launch_bounds__(256)
void embed_kernel(const float* __restrict__ x, const float* __restrict__ cont_w,
                  const float* __restrict__ cont_b, const float* __restrict__ ln_g,
                  const float* __restrict__ ln_b, const float* __restrict__ dir_emb,
                  float* __restrict__ hf, float* __restrict__ hb) {
    const int m = blockIdx.x;
    const int d = threadIdx.x;
    const int b = m >> 10, t = m & 1023;
    __shared__ float red[256];
    const float* xr = x + (size_t)m * 6;
    float e = cont_b[d];
    e += cont_w[d * 5 + 0] * xr[0];
    e += cont_w[d * 5 + 1] * xr[1];
    e += cont_w[d * 5 + 2] * xr[3];
    e += cont_w[d * 5 + 3] * xr[4];
    e += cont_w[d * 5 + 4] * xr[5];
    red[d] = e; __syncthreads();
    for (int s = 128; s > 0; s >>= 1) { if (d < s) red[d] += red[d + s]; __syncthreads(); }
    float mu = red[0] * (1.0f / 256.0f); __syncthreads();
    float c = e - mu;
    red[d] = c * c; __syncthreads();
    for (int s = 128; s > 0; s >>= 1) { if (d < s) red[d] += red[d + s]; __syncthreads(); }
    float var = red[0] * (1.0f / 256.0f);
    float xn = c * rsqrtf(var + 1e-5f) * ln_g[d] + ln_b[d];
    float g  = 0.5f * xn * (1.0f + erff(xn * 0.70710678f));
    int dir  = (xr[2] > 0.0f) ? 1 : 0;
    float v  = g + dir_emb[dir * 256 + d];
    float div = __expf(-(float)(d & ~1) * (9.21034037f / 256.0f));
    float ang = (float)t * div;
    v += (d & 1) ? __cosf(ang) : __sinf(ang);
    hf[(size_t)m * 256 + d] = v;
    hb[((size_t)b * 1024 + (1023 - t)) * 256 + d] = v;
}

// ---------------------------------------------------------------------------
// Row LayerNorm -> bf16
// ---------------------------------------------------------------------------
__global__ __launch_bounds__(256)
void ln_bf16_kernel(const float* __restrict__ x, const float* __restrict__ g,
                    const float* __restrict__ bta, __bf16* __restrict__ out) {
    const int m = blockIdx.x, d = threadIdx.x;
    __shared__ float red[256];
    float v = x[(size_t)m * 256 + d];
    red[d] = v; __syncthreads();
    for (int s = 128; s > 0; s >>= 1) { if (d < s) red[d] += red[d + s]; __syncthreads(); }
    float mu = red[0] * (1.0f / 256.0f); __syncthreads();
    float c = v - mu;
    red[d] = c * c; __syncthreads();
    for (int s = 128; s > 0; s >>= 1) { if (d < s) red[d] += red[d + s]; __syncthreads(); }
    float var = red[0] * (1.0f / 256.0f);
    out[(size_t)m * 256 + d] = f2bf(c * rsqrtf(var + 1e-5f) * g[d] + bta[d]);
}

// ---------------------------------------------------------------------------
// WMMA GEMM: C[MxN](f32) = A[MxK](bf16) * W[NxK](bf16)^T
// 256 threads = 8 wave32, block tile 128 x (NT*32), K stepped by 32.
// A tile: per-lane global_load_async_to_lds_b128 (ASYNCcnt).
// W tile: one TDM tensor_load_to_lds per step from wave 0 (TENSORcnt); the
//         TDM pad feature (16-DWORD interval, 4-DWORD pad) reproduces the
//         LDK=40 padded row stride in LDS. Fallback: async path.
// ---------------------------------------------------------------------------
template <int MODE, int NT>
__global__ __launch_bounds__(256)
void gemm_wmma(const __bf16* __restrict__ A, const __bf16* __restrict__ W,
               float* __restrict__ C, const float* __restrict__ Res,
               int M, int N, int K) {
    constexpr int BM = 128, BN = NT * 32, BK = 32, LDK = BK + 8;
    __shared__ __bf16 As[2][BM][LDK];
    __shared__ __bf16 Ws[2][BN][LDK];
    const int tid  = threadIdx.x;
    const int wave = tid >> 5, lane = tid & 31;
    const int waveM = wave >> 1, waveN = wave & 1;
    const int m0 = blockIdx.y * BM, n0 = blockIdx.x * BN;
    const int laneHalf = lane >> 4, lane15 = lane & 15;
    const int kbase = laneHalf * 8;

    floatx8 acc[2][NT] = {};

    const int arow = tid >> 2, aseg = tid & 3;       // 64 rows x 4 segs per pass

    auto stageA = [&](int bb, int k0) {
        ASYNC_LDCP(&As[bb][arow][aseg * 8],
                   A + (size_t)(m0 + arow) * K + k0 + aseg * 8);
        ASYNC_LDCP(&As[bb][arow + 64][aseg * 8],
                   A + (size_t)(m0 + arow + 64) * K + k0 + aseg * 8);
    };

    auto stageW = [&](int bb, int k0) {
#ifdef HAVE_TDM
        if (wave == 0) {
            unsigned lds = (unsigned)(size_t)&Ws[bb][0][0];
            unsigned long long ga =
                (unsigned long long)(size_t)(W + (size_t)n0 * K + k0);
            v4u g0;
            g0.x = 1u;                                        // count=1, normal mode
            g0.y = lds;                                       // lds_addr
            g0.z = (unsigned)(ga & 0xffffffffu);              // global_addr lo
            g0.w = (unsigned)((ga >> 32) & 0x01ffffffu) | (2u << 30); // hi | type=2
            v8i g1;
            g1[0] = (int)((1u << 16)        // data_size = 2 bytes
                        | (1u << 20)        // pad_enable
                        | (3u << 22)        // pad_interval = 16 DWORDs (one 64B row)
                        | (3u << 25));      // pad_amount  = 4 DWORDs (16B) -> LDK=40
            g1[1] = (int)(((unsigned)K & 0xffffu) << 16);     // tensor_dim0 lo16
            g1[2] = (int)(((unsigned)K >> 16) |
                          (((unsigned)N & 0xffffu) << 16));   // dim0 hi / dim1 lo
            g1[3] = (int)(((unsigned)N >> 16) | (32u << 16)); // dim1 hi / tile_dim0=32
            g1[4] = BN;                                       // tile_dim1=BN, tile_dim2=0
            g1[5] = K;                                        // tensor_dim0_stride lo32
            g1[6] = 0;
            g1[7] = 0;
            v4i z4 = {0, 0, 0, 0};                            // 2D tensor: groups 2/3 unused
            v8i z8 = {0, 0, 0, 0, 0, 0, 0, 0};
            __builtin_amdgcn_tensor_load_to_lds(g0, g1, z4, z4, z8, 0);
        }
#else
        ASYNC_LDCP(&Ws[bb][arow][aseg * 8],
                   W + (size_t)(n0 + arow) * K + k0 + aseg * 8);
        if (NT == 4)
            ASYNC_LDCP(&Ws[bb][arow + 64][aseg * 8],
                       W + (size_t)(n0 + arow + 64) * K + k0 + aseg * 8);
#endif
    };

    stageA(0, 0);
    stageW(0, 0);
    int cur = 0;
    for (int k0 = 0; k0 < K; k0 += BK) {
        WAIT_ASYNC();          // my per-lane async A copies have landed
        WAIT_TENSOR();         // wave0: W tile DMA complete (no-op elsewhere)
        __syncthreads();       // publish LDS tile to all waves
        if (k0 + BK < K) {     // kick next tile's DMA during the math
            stageA(cur ^ 1, k0 + BK);
            stageW(cur ^ 1, k0 + BK);
        }

        bf16x16 afrag[2], wfrag[NT];
        #pragma unroll
        for (int mt = 0; mt < 2; ++mt) {
            const __bf16* src = &As[cur][waveM * 32 + mt * 16 + lane15][kbase];
            ((uint4*)&afrag[mt])[0] = *(const uint4*)src;
            ((uint4*)&afrag[mt])[1] = *(const uint4*)(src + 16);
        }
        #pragma unroll
        for (int nt = 0; nt < NT; ++nt) {
            const __bf16* src = &Ws[cur][waveN * NT * 16 + nt * 16 + lane15][kbase];
            ((uint4*)&wfrag[nt])[0] = *(const uint4*)src;
            ((uint4*)&wfrag[nt])[1] = *(const uint4*)(src + 16);
        }
        #pragma unroll
        for (int mt = 0; mt < 2; ++mt)
            #pragma unroll
            for (int nt = 0; nt < NT; ++nt)
                acc[mt][nt] = __builtin_amdgcn_wmma_f32_16x16x32_bf16(
                    false, afrag[mt], false, wfrag[nt],
                    (short)0, acc[mt][nt], false, false);
        cur ^= 1;
    }

    #pragma unroll
    for (int mt = 0; mt < 2; ++mt) {
        #pragma unroll
        for (int nt = 0; nt < NT; ++nt) {
            int col = n0 + waveN * NT * 16 + nt * 16 + lane15;
            #pragma unroll
            for (int r = 0; r < 8; ++r) {
                int rowm = m0 + waveM * 32 + mt * 16 + r + laneHalf * 8;
                size_t off = (size_t)rowm * N + col;
                float v = acc[mt][nt][r];
                if (MODE == 1) v += Res[off];
                C[off] = v;
            }
        }
    }
}

// ---------------------------------------------------------------------------
// Causal depthwise conv (K=4) + bias + SiLU; writes f32 and bf16 copies.
// ---------------------------------------------------------------------------
__global__ __launch_bounds__(256)
void conv_silu_kernel(const float* __restrict__ xz, const float* __restrict__ cw,
                      const float* __restrict__ cb,
                      float* __restrict__ xi, __bf16* __restrict__ xib) {
    size_t idx = (size_t)blockIdx.x * 256 + threadIdx.x;   // M*512
    int d = (int)(idx & 511);
    size_t m = idx >> 9;
    int t = (int)(m & 1023);
    float acc = cb[d];
    #pragma unroll
    for (int k = 0; k < 4; ++k) {
        int tt = t - 3 + k;
        if (tt >= 0) acc += cw[d * 4 + k] * xz[(m - 3 + k) * 1024 + d];
    }
    float s = acc / (1.0f + __expf(-acc));
    xi[idx]  = s;
    xib[idx] = f2bf(s);
}

// ---------------------------------------------------------------------------
// Chunk-parallel selective scan (h_t = a_t h_{t-1} + w_t is associative).
// ---------------------------------------------------------------------------
__global__ __launch_bounds__(512)
void scan_p1(const float* __restrict__ xi, const float* __restrict__ xdbl,
             const float* __restrict__ dt_w, const float* __restrict__ dt_b,
             const float* __restrict__ A_log,
             float* __restrict__ Ac, float* __restrict__ Hc) {
    const int b  = blockIdx.x >> 5;
    const int c  = blockIdx.x & 31;
    const int di = threadIdx.x;
    __shared__ float s48[48];
    float A[SS], dw[DTR], a[SS], h[SS];
    #pragma unroll
    for (int s = 0; s < SS; ++s) { A[s] = -__expf(A_log[di * SS + s]); a[s] = 1.0f; h[s] = 0.0f; }
    #pragma unroll
    for (int r = 0; r < DTR; ++r) dw[r] = dt_w[di * DTR + r];
    const float dtb = dt_b[di];

    for (int tt = 0; tt < CLEN; ++tt) {
        size_t row = (size_t)b * LL + c * CLEN + tt;
        __syncthreads();
        if (di < 48) s48[di] = xdbl[row * 64 + di];
        __syncthreads();
        float pre = dtb;
        #pragma unroll
        for (int r = 0; r < DTR; ++r) pre += s48[r] * dw[r];
        float dt = (pre > 20.0f) ? pre : log1pf(__expf(pre));
        float du = dt * xi[row * 512 + di];
        #pragma unroll
        for (int s = 0; s < SS; ++s) {
            float dA = __expf(dt * A[s]);
            a[s] *= dA;
            h[s]  = dA * h[s] + du * s48[16 + s];
        }
    }
    size_t o = (((size_t)b * NCH + c) * 512 + di) * SS;
    #pragma unroll
    for (int s = 0; s < SS; ++s) { Ac[o + s] = a[s]; Hc[o + s] = h[s]; }
}

__global__ __launch_bounds__(256)
void scan_p2(const float* __restrict__ Ac, const float* __restrict__ Hc,
             float* __restrict__ Hin) {
    int i = blockIdx.x * 256 + threadIdx.x;     // BB * 512 * SS
    int b  = i >> 13;
    int ds = i & 8191;
    float h = 0.0f;
    for (int c = 0; c < NCH; ++c) {
        size_t o = ((size_t)b * NCH + c) * 8192 + ds;
        Hin[o] = h;
        h = Ac[o] * h + Hc[o];
    }
}

__global__ __launch_bounds__(512)
void scan_p3(const float* __restrict__ xi, const float* __restrict__ xdbl,
             const float* __restrict__ xz, const float* __restrict__ dt_w,
             const float* __restrict__ dt_b, const float* __restrict__ A_log,
             const float* __restrict__ Dp, const float* __restrict__ Hin,
             __bf16* __restrict__ yg) {
    const int b  = blockIdx.x >> 5;
    const int c  = blockIdx.x & 31;
    const int di = threadIdx.x;
    __shared__ float s48[48];
    float A[SS], dw[DTR], h[SS];
    #pragma unroll
    for (int s = 0; s < SS; ++s) A[s] = -__expf(A_log[di * SS + s]);
    #pragma unroll
    for (int r = 0; r < DTR; ++r) dw[r] = dt_w[di * DTR + r];
    const float dtb = dt_b[di], Dv = Dp[di];
    size_t o = (((size_t)b * NCH + c) * 512 + di) * SS;
    #pragma unroll
    for (int s = 0; s < SS; ++s) h[s] = Hin[o + s];

    for (int tt = 0; tt < CLEN; ++tt) {
        size_t row = (size_t)b * LL + c * CLEN + tt;
        __syncthreads();
        if (di < 48) s48[di] = xdbl[row * 64 + di];
        __syncthreads();
        float pre = dtb;
        #pragma unroll
        for (int r = 0; r < DTR; ++r) pre += s48[r] * dw[r];
        float dt = (pre > 20.0f) ? pre : log1pf(__expf(pre));
        float u  = xi[row * 512 + di];
        float du = dt * u;
        float y  = 0.0f;
        #pragma unroll
        for (int s = 0; s < SS; ++s) {
            float dA = __expf(dt * A[s]);
            h[s] = dA * h[s] + du * s48[16 + s];
            y   += h[s] * s48[32 + s];
        }
        y += u * Dv;
        float z = xz[row * 1024 + 512 + di];
        yg[row * 512 + di] = f2bf(y * (z / (1.0f + __expf(-z))));
    }
}

// ---------------------------------------------------------------------------
// Mean-pool over L (flip-invariant) + final projection
// ---------------------------------------------------------------------------
__global__ __launch_bounds__(256)
void pool_kernel(const float* __restrict__ hf, const float* __restrict__ hb,
                 float* __restrict__ zc) {
    int b = blockIdx.x, d = threadIdx.x;
    float s1 = 0.0f, s2 = 0.0f;
    for (int t = 0; t < LL; ++t) {
        s1 += hf[((size_t)b * LL + t) * 256 + d];
        s2 += hb[((size_t)b * LL + t) * 256 + d];
    }
    zc[b * 512 + d]       = s1 * (1.0f / 1024.0f);
    zc[b * 512 + 256 + d] = s2 * (1.0f / 1024.0f);
}

__global__ __launch_bounds__(128)
void proj_kernel(const float* __restrict__ zc, const float* __restrict__ pw,
                 const float* __restrict__ pb, float* __restrict__ out) {
    int b = blockIdx.x, o = threadIdx.x;
    float acc = pb[o];
    const float* z = zc + b * 512;
    const float* w = pw + (size_t)o * 512;
    for (int k = 0; k < 512; ++k) acc += z[k] * w[k];
    out[b * OUTD + o] = acc;
}

// ---------------------------------------------------------------------------
// Host launcher
// ---------------------------------------------------------------------------
extern "C" void kernel_launch(void* const* d_in, const int* in_sizes, int n_in,
                              void* d_out, int out_size, void* d_ws, size_t ws_size,
                              hipStream_t stream) {
    const float* x       = (const float*)d_in[0];
    const float* cont_w  = (const float*)d_in[1];
    const float* cont_b  = (const float*)d_in[2];
    const float* ln_g    = (const float*)d_in[3];
    const float* ln_b    = (const float*)d_in[4];
    const float* dir_emb = (const float*)d_in[5];
    const float* in_w    = (const float*)d_in[6];
    const float* conv_w  = (const float*)d_in[7];
    const float* conv_b  = (const float*)d_in[8];
    const float* xproj_w = (const float*)d_in[9];
    const float* dt_w    = (const float*)d_in[10];
    const float* dt_b    = (const float*)d_in[11];
    const float* A_log   = (const float*)d_in[12];
    const float* Dp      = (const float*)d_in[13];
    const float* out_w   = (const float*)d_in[14];
    const float* norm_g  = (const float*)d_in[15];
    const float* norm_b  = (const float*)d_in[16];
    const float* proj_w  = (const float*)d_in[17];
    const float* proj_b  = (const float*)d_in[18];

    char* base = (char*)d_ws;
    size_t off = 0;
    auto carve = [&](size_t bytes) -> void* {
        void* p = base + off;
        off += (bytes + 255) & ~(size_t)255;
        return p;
    };
    __bf16* xn    = (__bf16*)carve((size_t)MM * DD * 2);
    float*  xz    = (float*) carve((size_t)MM * 1024 * 4);
    float*  xi    = (float*) carve((size_t)MM * DI * 4);
    __bf16* xib   = (__bf16*)carve((size_t)MM * DI * 2);
    float*  xdbl  = (float*) carve((size_t)MM * 64 * 4);
    __bf16* ygb   = (__bf16*)carve((size_t)MM * DI * 2);
    float*  hf    = (float*) carve((size_t)MM * DD * 4);
    float*  hb    = (float*) carve((size_t)MM * DD * 4);
    float*  Ac    = (float*) carve((size_t)BB * NCH * DI * SS * 4);
    float*  Hc    = (float*) carve((size_t)BB * NCH * DI * SS * 4);
    float*  Hin   = (float*) carve((size_t)BB * NCH * DI * SS * 4);
    __bf16* w_in  = (__bf16*)carve((size_t)NL * 1024 * 256 * 2);
    __bf16* w_out = (__bf16*)carve((size_t)NL * 256 * 512 * 2);
    __bf16* w_xp  = (__bf16*)carve((size_t)NL * 64 * 512 * 2);
    float*  zc    = (float*) carve((size_t)BB * 512 * 4);

    cvt_bf16_kernel<<<(NL * 1024 * 256 + 255) / 256, 256, 0, stream>>>(in_w, w_in, NL * 1024 * 256);
    cvt_bf16_kernel<<<(NL * 256 * 512 + 255) / 256, 256, 0, stream>>>(out_w, w_out, NL * 256 * 512);
    cvt_xproj_kernel<<<(NL * 64 * 512) / 256, 256, 0, stream>>>(xproj_w, w_xp);

    embed_kernel<<<MM, 256, 0, stream>>>(x, cont_w, cont_b, ln_g, ln_b, dir_emb, hf, hb);

    for (int l = 0; l < NL; ++l) {
        float* xcur = (l < 2) ? hf : hb;
        ln_bf16_kernel<<<MM, 256, 0, stream>>>(xcur, norm_g + l * 256, norm_b + l * 256, xn);
        // in-proj: (32768x256)x(256x1024), block tile 128x128
        gemm_wmma<0, 4><<<dim3(1024 / 128, MM / 128), 256, 0, stream>>>(
            xn, w_in + (size_t)l * 1024 * 256, xz, nullptr, MM, 1024, 256);
        conv_silu_kernel<<<(MM * DI) / 256, 256, 0, stream>>>(
            xz, conv_w + (size_t)l * 512 * 4, conv_b + (size_t)l * 512, xi, xib);
        // x-proj (N padded to 64): block tile 128x64
        gemm_wmma<0, 2><<<dim3(1, MM / 128), 256, 0, stream>>>(
            xib, w_xp + (size_t)l * 64 * 512, xdbl, nullptr, MM, 64, 512);
        // chunk-parallel selective scan
        scan_p1<<<BB * NCH, 512, 0, stream>>>(
            xi, xdbl, dt_w + (size_t)l * 512 * 16, dt_b + (size_t)l * 512,
            A_log + (size_t)l * 512 * 16, Ac, Hc);
        scan_p2<<<(BB * 512 * SS) / 256, 256, 0, stream>>>(Ac, Hc, Hin);
        scan_p3<<<BB * NCH, 512, 0, stream>>>(
            xi, xdbl, xz, dt_w + (size_t)l * 512 * 16, dt_b + (size_t)l * 512,
            A_log + (size_t)l * 512 * 16, Dp + (size_t)l * 512, Hin, ygb);
        // out-proj + residual: (32768x512)x(512x256), block tile 128x128
        gemm_wmma<1, 4><<<dim3(256 / 128, MM / 128), 256, 0, stream>>>(
            ygb, w_out + (size_t)l * 256 * 512, xcur, xcur, MM, 256, 512);
    }

    pool_kernel<<<BB, 256, 0, stream>>>(hf, hb, zc);
    proj_kernel<<<BB, 128, 0, stream>>>(zc, proj_w, proj_b, (float*)d_out);
}
